// TemporalAttention_29515015258330
// MI455X (gfx1250) — matrix operand order repaired and checked by reading, not compiled, not run on hardware
//
#include <hip/hip_runtime.h>
#include <hip/hip_bf16.h>

// ---------------------------------------------------------------------------
// Transformer block (QKV proj -> 12-head attention -> FC+bias+residual -> LN)
// MI455X / gfx1250: wave32, bf16 WMMA (v_wmma_f32_16x16x32_bf16) everywhere.
// Attention streams K/V chunks into LDS with async global->LDS loads
// (ASYNCcnt), double-buffered, shared by all 8 waves of the block.
// ---------------------------------------------------------------------------

#define D_MODEL 768
#define N_HEADS 12
#define D_K     64
#define SEQ     2048
#define BATCH   4
#define N_TOK   (BATCH * SEQ)   // 8192 rows

typedef __attribute__((ext_vector_type(16))) __bf16 v16bf;
typedef __attribute__((ext_vector_type(8)))  float  v8f;
typedef __attribute__((ext_vector_type(4)))  int    v4i;

#if __has_builtin(__builtin_amdgcn_global_load_async_to_lds_b128) && \
    __has_builtin(__builtin_amdgcn_s_wait_asynccnt)
#define USE_ASYNC_LDS 1
#else
#define USE_ASYNC_LDS 0
#endif

__device__ __forceinline__ unsigned short f2bf(float f) {
  unsigned int u = __float_as_uint(f);
  u += 0x7FFFu + ((u >> 16) & 1u);     // round-to-nearest-even
  return (unsigned short)(u >> 16);
}

__device__ __forceinline__ __bf16 us2bf(unsigned short u) {
  __bf16 h; __builtin_memcpy(&h, &u, 2); return h;
}

// 16-byte global -> LDS copy. Async (ASYNCcnt-tracked) when available.
// Builtin signature (from hipcc diagnostic): param1 = int4 __device__* (as1),
// param2 = int4 LDS ptr (as3), then imm offset + imm cpol.
__device__ __forceinline__ void copy16_g2l(const unsigned short* g,
                                           unsigned short* l) {
#if USE_ASYNC_LDS
  __builtin_amdgcn_global_load_async_to_lds_b128(
      (__attribute__((address_space(1))) v4i*)g,
      (__attribute__((address_space(3))) v4i*)l, 0, 0);
#else
  *(uint4*)l = *(const uint4*)g;
#endif
}
__device__ __forceinline__ void wait_async_2() {
#if USE_ASYNC_LDS
  __builtin_amdgcn_s_wait_asynccnt(2);   // keep the 2 just-issued in flight
#endif
}
__device__ __forceinline__ void wait_async_0() {
#if USE_ASYNC_LDS
  __builtin_amdgcn_s_wait_asynccnt(0);
#endif
}

// A-matrix 16x32 bf16 fragment strip (ISA 7.12.2): lane = row (lane&15),
// element e -> K = g*8 + e + 8*(e>>3), g = lane>>4.  st = element stride.
__device__ __forceinline__ v16bf ld_strip_a(const unsigned short* p, int st, int g) {
  v16bf r;
#pragma unroll
  for (int e = 0; e < 16; ++e) {
    int k = g * 8 + e + ((e >> 3) << 3);
    r[e] = us2bf(p[k * st]);
  }
  return r;
}

// B-matrix 32x16 bf16 fragment strip: lane = column (lane&15),
// element e -> K = g*8 + e + 8*(e>>3).
__device__ __forceinline__ v16bf ld_strip_b(const unsigned short* p, int st, int g) {
  v16bf r;
#pragma unroll
  for (int e = 0; e < 16; ++e) {
    int k = g * 8 + e + ((e >> 3) << 3);
    r[e] = us2bf(p[k * st]);
  }
  return r;
}

__device__ __forceinline__ v8f wmma_bf16(v16bf a, v16bf b, v8f c) {
  return __builtin_amdgcn_wmma_f32_16x16x32_bf16(false, a, false, b,
                                                 (short)0, c, false, false);
}

// ---------------------------------------------------------------------------
// Stage 0: f32 -> bf16 conversion
// ---------------------------------------------------------------------------
__global__ void cvt_f32_bf16(const float* __restrict__ in,
                             unsigned short* __restrict__ out, int n) {
  int i = blockIdx.x * blockDim.x + threadIdx.x;
  if (i < n) out[i] = f2bf(in[i]);
}

// ---------------------------------------------------------------------------
// Stage 1: fused QKV projection.  Y = X * W^T  (W stored [out,in]).
//   Q -> [B,H,S,Dk] bf16, pre-scaled by 1/sqrt(Dk)
//   K -> [B,H,S,Dk] bf16
//   V -> [B,H,Dk,S] bf16 (transposed, so PV fragment loads are contiguous)
// ---------------------------------------------------------------------------
__global__ void __launch_bounds__(256)
qkv_gemm(const unsigned short* __restrict__ xb,
         const unsigned short* __restrict__ wq,
         const unsigned short* __restrict__ wk,
         const unsigned short* __restrict__ wv,
         unsigned short* __restrict__ Qb,
         unsigned short* __restrict__ Kb,
         unsigned short* __restrict__ Vtb) {
  const int wave = (blockIdx.x * blockDim.x + threadIdx.x) >> 5;
  const int lane = threadIdx.x & 31;
  const int r16 = lane & 15, g = lane >> 4;
  const int TPN = D_MODEL / 16;                 // 48 tiles along N
  const int tilesPerMat = (N_TOK / 16) * TPN;   // 24576
  const int mat = wave / tilesPerMat;
  const int t   = wave % tilesPerMat;
  const int m0 = (t / TPN) * 16, n0 = (t % TPN) * 16;

  const unsigned short* w = (mat == 0) ? wq : (mat == 1) ? wk : wv;
  const unsigned short* pa = xb + (size_t)(m0 + r16) * D_MODEL;
  const unsigned short* pb = w  + (size_t)(n0 + r16) * D_MODEL;

  v8f c = {};
#pragma unroll 4
  for (int k0 = 0; k0 < D_MODEL; k0 += 32) {
    v16bf a = ld_strip_a(pa + k0, 1, g);
    v16bf b = ld_strip_b(pb + k0, 1, g);
    c = wmma_bf16(a, b, c);
  }

#pragma unroll
  for (int r = 0; r < 8; ++r) {
    int m = m0 + r + 8 * g;            // D-layout: VGPR r, half g -> row
    int n = n0 + r16;                  // lane -> column
    int bb = m / SEQ, s = m % SEQ;
    int h = n / D_K, d = n % D_K;
    if (mat == 2) {
      Vtb[(((size_t)(bb * N_HEADS + h) * D_K) + d) * SEQ + s] = f2bf(c[r]);
    } else {
      float v = (mat == 0) ? c[r] * 0.125f : c[r];   // fold 1/sqrt(64) into Q
      unsigned short* dst = (mat == 0) ? Qb : Kb;
      dst[(((size_t)(bb * N_HEADS + h) * SEQ) + s) * D_K + d] = f2bf(v);
    }
  }
}

// ---------------------------------------------------------------------------
// Stage 2: flash attention.  A block owns 8 consecutive q-tiles of one (b,h);
// its 8 waves share double-buffered K/V chunks staged in LDS via async
// global->LDS loads.  One wave = one 16-row q-tile, online softmax over
// 32-key chunks, P staged through LDS to become the PV A-fragment.
// ---------------------------------------------------------------------------
__global__ void __launch_bounds__(256)
flash_attn(const unsigned short* __restrict__ Qb,
           const unsigned short* __restrict__ Kb,
           const unsigned short* __restrict__ Vtb,
           unsigned short* __restrict__ ctxb) {
  __shared__ alignas(16) unsigned short kbuf[2][32 * D_K];  // [key][d]  4KB each
  __shared__ alignas(16) unsigned short vbuf[2][D_K * 32];  // [d][key]  4KB each
  __shared__ alignas(16) unsigned short lds_p[8][16 * 32];  // per-wave P tile
  const int tid   = threadIdx.x;
  const int wslot = tid >> 5;
  const int lane  = tid & 31;
  const int r16 = lane & 15, g = lane >> 4;

  const int bh = blockIdx.x >> 4;                       // 48 (b,h) pairs
  const int q0 = ((blockIdx.x & 15) * 8 + wslot) * 16;  // wave's q-tile
  const unsigned short* Qh  = Qb  + (size_t)bh * SEQ * D_K;
  const unsigned short* Kh  = Kb  + (size_t)bh * SEQ * D_K;
  const unsigned short* Vth = Vtb + (size_t)bh * D_K * SEQ;

  // Cooperative chunk staging: 2 x 16B per thread (4KB K + 4KB V per chunk).
  const int vd = tid >> 2, vp = tid & 3;
  auto stage = [&](int nbuf, int j0) {
    copy16_g2l(Kh + (size_t)j0 * D_K + tid * 8, &kbuf[nbuf][tid * 8]);
    copy16_g2l(Vth + (size_t)vd * SEQ + j0 + vp * 8, &vbuf[nbuf][vd * 32 + vp * 8]);
  };

  const unsigned short* pq = Qh + (size_t)(q0 + r16) * D_K;
  v16bf aq0 = ld_strip_a(pq, 1, g);        // Q d=0..31 (already * 1/sqrt(Dk))
  v16bf aq1 = ld_strip_a(pq + 32, 1, g);   // Q d=32..63

  v8f o0 = {}, o1 = {}, o2 = {}, o3 = {};  // 16x64 fp32 ctx accumulator
  float mrow[8], lrow[8];
#pragma unroll
  for (int r = 0; r < 8; ++r) { mrow[r] = -1e30f; lrow[r] = 0.f; }

  stage(0, 0);                             // preload first chunk
  for (int j0 = 0; j0 < SEQ; j0 += 32) {
    const int buf = (j0 >> 5) & 1;
    if (j0 + 32 < SEQ) {                   // uniform branch
      stage(buf ^ 1, j0 + 32);             // stream next chunk into other buffer
      wait_async_2();                      // current chunk's loads complete
    } else {
      wait_async_0();
    }
    __syncthreads();                       // (A) chunk visible to all waves

    // --- scores: Q(16x64) * K^T from LDS --------------------------------
    const unsigned short* pk0 = &kbuf[buf][(size_t)r16 * D_K];
    const unsigned short* pk1 = &kbuf[buf][(size_t)(16 + r16) * D_K];
    v8f s0 = {}, s1 = {};
    s0 = wmma_bf16(aq0, ld_strip_b(pk0, 1, g), s0);
    s0 = wmma_bf16(aq1, ld_strip_b(pk0 + 32, 1, g), s0);
    s1 = wmma_bf16(aq0, ld_strip_b(pk1, 1, g), s1);
    s1 = wmma_bf16(aq1, ld_strip_b(pk1 + 32, 1, g), s1);

    // --- online softmax update (rows live across 16 lanes of a half) ----
#pragma unroll
    for (int r = 0; r < 8; ++r) {
      float sv0 = s0[r];
      float sv1 = s1[r];
      float mx = fmaxf(sv0, sv1);
#pragma unroll
      for (int msk = 1; msk < 16; msk <<= 1)
        mx = fmaxf(mx, __shfl_xor(mx, msk, 32));
      float mnew = fmaxf(mrow[r], mx);
      float scl  = __expf(mrow[r] - mnew);
      float p0 = __expf(sv0 - mnew);
      float p1 = __expf(sv1 - mnew);
      float ps = p0 + p1;
#pragma unroll
      for (int msk = 1; msk < 16; msk <<= 1)
        ps += __shfl_xor(ps, msk, 32);
      lrow[r] = lrow[r] * scl + ps;
      mrow[r] = mnew;
      o0[r] *= scl; o1[r] *= scl; o2[r] *= scl; o3[r] *= scl;
      lds_p[wslot][(r + 8 * g) * 32 + r16]      = f2bf(p0);
      lds_p[wslot][(r + 8 * g) * 32 + 16 + r16] = f2bf(p1);
    }
    __syncthreads();                       // (B) P tile committed

    // --- PV: P(16x32) * V(32x64) from LDS --------------------------------
    v16bf ap = ld_strip_a(&lds_p[wslot][r16 * 32], 1, g);
    const unsigned short* pv = &vbuf[buf][(size_t)r16 * 32];
    o0 = wmma_bf16(ap, ld_strip_b(pv,           1, g), o0);
    o1 = wmma_bf16(ap, ld_strip_b(pv + 16 * 32, 1, g), o1);
    o2 = wmma_bf16(ap, ld_strip_b(pv + 32 * 32, 1, g), o2);
    o3 = wmma_bf16(ap, ld_strip_b(pv + 48 * 32, 1, g), o3);
    __syncthreads();                       // (C) chunk fully consumed
  }

  // --- normalize and write ctx back in [B,S,H*Dk] bf16 layout -------------
  const int b = bh / N_HEADS, h = bh % N_HEADS;
#pragma unroll
  for (int r = 0; r < 8; ++r) {
    float inv = 1.f / lrow[r];
    int srow = q0 + r + 8 * g;
    unsigned short* pc = ctxb + ((size_t)(b * SEQ + srow)) * D_MODEL + h * D_K + r16;
    pc[0]  = f2bf(o0[r] * inv);
    pc[16] = f2bf(o1[r] * inv);
    pc[32] = f2bf(o2[r] * inv);
    pc[48] = f2bf(o3[r] * inv);
  }
}

// ---------------------------------------------------------------------------
// Stage 3: output projection + bias + residual (fp32 out into d_out)
// ---------------------------------------------------------------------------
__global__ void __launch_bounds__(256)
fc_gemm(const unsigned short* __restrict__ ctxb,
        const unsigned short* __restrict__ wfc,
        const float* __restrict__ bias,
        const float* __restrict__ x,
        float* __restrict__ out) {
  const int wave = (blockIdx.x * blockDim.x + threadIdx.x) >> 5;
  const int lane = threadIdx.x & 31;
  const int r16 = lane & 15, g = lane >> 4;
  const int TPN = D_MODEL / 16;
  const int m0 = (wave / TPN) * 16, n0 = (wave % TPN) * 16;

  const unsigned short* pa = ctxb + (size_t)(m0 + r16) * D_MODEL;
  const unsigned short* pb = wfc  + (size_t)(n0 + r16) * D_MODEL;
  v8f c = {};
#pragma unroll 4
  for (int k0 = 0; k0 < D_MODEL; k0 += 32)
    c = wmma_bf16(ld_strip_a(pa + k0, 1, g), ld_strip_b(pb + k0, 1, g), c);

#pragma unroll
  for (int r = 0; r < 8; ++r) {
    int m = m0 + r + 8 * g, n = n0 + r16;
    out[(size_t)m * D_MODEL + n] = c[r] + bias[n] + x[(size_t)m * D_MODEL + n];
  }
}

// ---------------------------------------------------------------------------
// Stage 4: in-place LayerNorm over each 768-wide row of d_out
// ---------------------------------------------------------------------------
__global__ void __launch_bounds__(256)
layernorm(float* __restrict__ y, const float* __restrict__ gamma,
          const float* __restrict__ beta) {
  __shared__ float ssum[256], ssq[256];
  float* p = y + (size_t)blockIdx.x * D_MODEL;
  float s = 0.f, q = 0.f;
  for (int c = threadIdx.x; c < D_MODEL; c += 256) {
    float v = p[c]; s += v; q += v * v;
  }
  ssum[threadIdx.x] = s; ssq[threadIdx.x] = q;
  __syncthreads();
  for (int off = 128; off > 0; off >>= 1) {
    if (threadIdx.x < off) {
      ssum[threadIdx.x] += ssum[threadIdx.x + off];
      ssq[threadIdx.x]  += ssq[threadIdx.x + off];
    }
    __syncthreads();
  }
  float mean = ssum[0] * (1.f / D_MODEL);
  float var  = ssq[0] * (1.f / D_MODEL) - mean * mean;
  float inv  = rsqrtf(var + 1e-5f);
  for (int c = threadIdx.x; c < D_MODEL; c += 256)
    p[c] = (p[c] - mean) * inv * gamma[c] + beta[c];
}

// ---------------------------------------------------------------------------
extern "C" void kernel_launch(void* const* d_in, const int* in_sizes, int n_in,
                              void* d_out, int out_size, void* d_ws, size_t ws_size,
                              hipStream_t stream) {
  const float* x     = (const float*)d_in[0];
  const float* w_q   = (const float*)d_in[1];
  const float* w_k   = (const float*)d_in[2];
  const float* w_v   = (const float*)d_in[3];
  const float* w_fc  = (const float*)d_in[4];
  const float* b_fc  = (const float*)d_in[5];
  const float* gamma = (const float*)d_in[6];
  const float* beta  = (const float*)d_in[7];
  float* out = (float*)d_out;

  // Workspace layout (bf16 / ushort elements), total ~64.5 MB
  const size_t NX = (size_t)N_TOK * D_MODEL;     // 6,291,456
  const size_t NW = (size_t)D_MODEL * D_MODEL;   //   589,824
  unsigned short* ws   = (unsigned short*)d_ws;
  unsigned short* xb   = ws;
  unsigned short* wqb  = xb  + NX;
  unsigned short* wkb  = wqb + NW;
  unsigned short* wvb  = wkb + NW;
  unsigned short* wfb  = wvb + NW;
  unsigned short* Qb   = wfb + NW;
  unsigned short* Kb   = Qb  + NX;
  unsigned short* Vtb  = Kb  + NX;
  unsigned short* ctxb = Vtb + NX;

  // Stage 0: downcast activations + weights to bf16
  cvt_f32_bf16<<<(int)((NX + 255) / 256), 256, 0, stream>>>(x,    xb,  (int)NX);
  cvt_f32_bf16<<<(int)((NW + 255) / 256), 256, 0, stream>>>(w_q,  wqb, (int)NW);
  cvt_f32_bf16<<<(int)((NW + 255) / 256), 256, 0, stream>>>(w_k,  wkb, (int)NW);
  cvt_f32_bf16<<<(int)((NW + 255) / 256), 256, 0, stream>>>(w_v,  wvb, (int)NW);
  cvt_f32_bf16<<<(int)((NW + 255) / 256), 256, 0, stream>>>(w_fc, wfb, (int)NW);

  // Stage 1: QKV.  3 mats * 512*48 tiles = 73728 waves / 8 per block.
  qkv_gemm<<<9216, 256, 0, stream>>>(xb, wqb, wkb, wvb, Qb, Kb, Vtb);

  // Stage 2: attention. 48 heads * 16 q-groups = 768 blocks of 8 waves.
  flash_attn<<<768, 256, 0, stream>>>(Qb, Kb, Vtb, ctxb);

  // Stage 3: FC + bias + residual. 512*48 = 24576 waves / 8 per block.
  fc_gemm<<<3072, 256, 0, stream>>>(ctxb, wfb, b_fc, x, out);

  // Stage 4: LayerNorm, one block per row.
  layernorm<<<N_TOK, 256, 0, stream>>>(out, gamma, beta);
}
